// CrossWindowAttention_38448547234071
// MI455X (gfx1250) — matrix operand order repaired and checked by reading, not compiled, not run on hardware
//
#include <hip/hip_runtime.h>

// ---------------------------------------------------------------------------
// CrossWindowAttention (Swin3D window cross-attn) for MI455X / gfx1250.
// B=256, N=343 (pad 352), DIM=192, H=6, D=32, NW=64.
// ~49 GFLOP vs ~250MB mandatory HBM traffic -> fuse attention (no [B,H,N,N]
// in HBM), all matmuls via v_wmma_f32_16x16x32_bf16. Additive tables padded
// to 352x352 so the softmax inner loop is branchless, and softmax row-sums
// are computed by an extra WMMA against an all-ones fragment (P x 1), which
// removes the serialized ds_swizzle sum-reduction chain entirely.
// ---------------------------------------------------------------------------

typedef __attribute__((ext_vector_type(16))) __bf16 v16bf;
typedef __attribute__((ext_vector_type(8)))  __bf16 v8bf;
typedef __attribute__((ext_vector_type(8)))  float  v8f;

#define BATCH 256
#define NTOK  343
#define NPAD  352          // 22 tiles of 16
#define CDIM  192
#define HEADS 6
#define HDIM  32
#define NWIN  64
#define ROWS  (BATCH * NTOK)      // 87808
#define ROWBLOCKS (ROWS / 32)     // 2744

static __device__ inline __bf16 f2bf(float f) {
  unsigned u = __float_as_uint(f);
  unsigned r = (u + 0x7FFFu + ((u >> 16) & 1u)) >> 16;
  unsigned short s = (unsigned short)r;
  return __builtin_bit_cast(__bf16, s);
}

static __device__ inline v8f v8f_zero() {
  v8f z = {0.f, 0.f, 0.f, 0.f, 0.f, 0.f, 0.f, 0.f};
  return z;
}

static __device__ inline v8f wmma_bf16(v16bf a, v16bf b, v8f c) {
  // D = A(16x32 bf16) * B(32x16 bf16) + C(16x16 f32)
  return __builtin_amdgcn_wmma_f32_16x16x32_bf16(false, a, false, b,
                                                 (short)0, c, false, false);
}

// Build a v16bf fragment from two 8-element contiguous chunks (16B each).
static __device__ inline v16bf load_frag(const __bf16* p0, const __bf16* p1) {
  union { v16bf v; v8bf h[2]; } u;
  u.h[0] = *reinterpret_cast<const v8bf*>(p0);
  u.h[1] = *reinterpret_cast<const v8bf*>(p1);
  return u.v;
}

// XOR lane swap via ds_swizzle (group-of-32 mode: new_lane = lane ^ XM).
// XM in {1,2,4,8} never crosses the 16-lane half boundary.
template <int XM>
static __device__ inline float swz_xor(float x) {
  return __int_as_float(__builtin_amdgcn_ds_swizzle(
      __float_as_int(x), 0x1F | (XM << 10)));
}
static __device__ inline float red_max16(float x) {
  x = fmaxf(x, swz_xor<1>(x));
  x = fmaxf(x, swz_xor<2>(x));
  x = fmaxf(x, swz_xor<4>(x));
  x = fmaxf(x, swz_xor<8>(x));
  return x;
}

// ---------------------------------------------------------------------------
// Prep kernels
// ---------------------------------------------------------------------------
__global__ void cvt_f32_to_bf16(const float* __restrict__ in,
                                __bf16* __restrict__ out, int n) {
  int i = blockIdx.x * blockDim.x + threadIdx.x;
  if (i < n) out[i] = f2bf(in[i]);
}

// bias_dense[h][n][m] = rpb_table[rpi[n,m], h]  (pads -> -1e30)
__global__ void build_bias(const float* __restrict__ table,
                           const int* __restrict__ rpi,
                           float* __restrict__ biasD) {
  int i = blockIdx.x * blockDim.x + threadIdx.x;
  int total = HEADS * NPAD * NPAD;
  if (i >= total) return;
  int h = i / (NPAD * NPAD);
  int rem = i - h * (NPAD * NPAD);
  int n = rem / NPAD, m = rem - n * NPAD;
  float v = -1.0e30f;
  if (n < NTOK && m < NTOK) v = table[rpi[n * NTOK + m] * HEADS + h];
  biasD[i] = v;
}

// mask_pad[w][n][m]: mask zero-padded to 352x352 (pad cols already -1e30 in
// bias_dense), so the attention inner loop needs no bounds checks.
__global__ void build_maskP(const float* __restrict__ mask,
                            float* __restrict__ maskP) {
  int i = blockIdx.x * blockDim.x + threadIdx.x;
  int total = NWIN * NPAD * NPAD;
  if (i >= total) return;
  int w = i / (NPAD * NPAD);
  int rem = i - w * (NPAD * NPAD);
  int n = rem / NPAD, m = rem - n * NPAD;
  float v = 0.f;
  if (n < NTOK && m < NTOK) v = mask[((size_t)w * NTOK + n) * NTOK + m];
  maskP[i] = v;
}

// Zero Q/K/V pad rows (n in [343,352)) so attention never sees poison.
__global__ void zero_pads(__bf16* __restrict__ q, __bf16* __restrict__ k,
                          __bf16* __restrict__ v) {
  int i = blockIdx.x * blockDim.x + threadIdx.x;
  int total = BATCH * HEADS * (NPAD - NTOK) * HDIM;
  if (i >= total) return;
  int per = (NPAD - NTOK) * HDIM;             // 288
  int bh = i / per;
  int r = i - bh * per;
  int n = NTOK + r / HDIM, d = r % HDIM;
  size_t off = ((size_t)bh * NPAD + n) * HDIM + d;
  __bf16 z = f2bf(0.f);
  q[off] = z; k[off] = z; v[off] = z;
}

// ---------------------------------------------------------------------------
// Fused QKV projection GEMM.  MODE 0: Q (NT=3, scale folded in).
//                             MODE 1: KV (NT=6, split into K/V buffers).
// Block = 256 threads (8 waves); 32 output rows per block; A staged in LDS.
// Output layout: [b, h, n(pad 352), d] bf16.
// ---------------------------------------------------------------------------
template <int NT, int MODE>
__global__ __launch_bounds__(256) void gemm_qkv_kernel(
    const float* __restrict__ x, const __bf16* __restrict__ wbf,
    const float* __restrict__ bias, __bf16* __restrict__ out0,
    __bf16* __restrict__ out1) {
  __shared__ __bf16 sA[32 * CDIM];
  const int rb = blockIdx.x;
  const int tid = threadIdx.x;
  const float* xrow = x + (size_t)rb * 32 * CDIM;
#pragma unroll
  for (int j = 0; j < 24; ++j) {
    int idx = tid + j * 256;
    sA[idx] = f2bf(xrow[idx]);
  }
  __syncthreads();

  const int lane = tid & 31, wid = tid >> 5;
  const int half = lane >> 4, ln = lane & 15;
  const int mt = wid & 1;
  const int ntbase = (wid >> 1) * NT;

  v8f acc[NT];
#pragma unroll
  for (int nt = 0; nt < NT; ++nt) acc[nt] = v8f_zero();

  const __bf16* arow = &sA[(mt * 16 + ln) * CDIM];
#pragma unroll
  for (int k = 0; k < 6; ++k) {            // K = 6 * 32 = 192
    v16bf a = load_frag(arow + k * 32 + 16 * half,
                        arow + k * 32 + 16 * half + 8);
#pragma unroll
    for (int nt = 0; nt < NT; ++nt) {
      const __bf16* wrow =
          wbf + (size_t)((ntbase + nt) * 16 + ln) * CDIM + k * 32 + half * 16;
      v16bf b = load_frag(wrow, wrow + 8);
      acc[nt] = wmma_bf16(a, b, acc[nt]);
    }
  }

#pragma unroll
  for (int nt = 0; nt < NT; ++nt) {
    int col = (ntbase + nt) * 16 + ln;
    float bv = bias[col];
#pragma unroll
    for (int i = 0; i < 8; ++i) {
      int r = rb * 32 + mt * 16 + i + 8 * half;
      int bb = r / NTOK, n = r - bb * NTOK;
      float v = acc[nt][i] + bv;
      if (MODE == 0) {
        v *= 0.17677669529663687f;  // d^-0.5 folded into Q
        int h = col >> 5, d = col & 31;
        out0[((size_t)(bb * HEADS + h) * NPAD + n) * HDIM + d] = f2bf(v);
      } else {
        if (col < CDIM) {
          int h = col >> 5, d = col & 31;
          out0[((size_t)(bb * HEADS + h) * NPAD + n) * HDIM + d] = f2bf(v);
        } else {
          int c2 = col - CDIM;
          int h = c2 >> 5, d = c2 & 31;
          out1[((size_t)(bb * HEADS + h) * NPAD + n) * HDIM + d] = f2bf(v);
        }
      }
    }
  }
}

// ---------------------------------------------------------------------------
// Fused attention: one workgroup per (b,h).  352 threads = 11 waves.
// K [352,32] row-major and V transposed [32,352] in LDS; each wave owns two
// 16-row query tiles, streams over 11 key blocks of 32 with online softmax.
// P repacked to A-fragment layout through per-wave LDS.  Row-sums (softmax
// denominator) accumulate through WMMA against an all-ones fragment, so only
// the row-max needs a ds_swizzle reduction.
// ---------------------------------------------------------------------------
__global__ __launch_bounds__(352) void attention_kernel(
    const __bf16* __restrict__ Qs, const __bf16* __restrict__ Ks,
    const __bf16* __restrict__ Vs, const float* __restrict__ biasD,
    const float* __restrict__ maskP, __bf16* __restrict__ O) {
  __shared__ __bf16 sK[NPAD * HDIM];      // [m][k]
  __shared__ __bf16 sVt[HDIM * NPAD];     // [d][m]
  __shared__ __bf16 sP[11][16 * 32];      // per-wave P repack scratch

  const int bid = blockIdx.x;
  const int b = bid / HEADS, h = bid - b * HEADS;
  const int w = b & (NWIN - 1);
  const __bf16* Qg = Qs + (size_t)bid * NPAD * HDIM;
  const __bf16* Kg = Ks + (size_t)bid * NPAD * HDIM;
  const __bf16* Vg = Vs + (size_t)bid * NPAD * HDIM;
  const float* bh = biasD + (size_t)h * NPAD * NPAD;
  const float* mw = maskP + (size_t)w * NPAD * NPAD;

  const int tid = threadIdx.x;
  {  // stage K row-major, V transposed
    int m = tid;  // 0..351
    const uint4* ksrc = reinterpret_cast<const uint4*>(Kg + m * HDIM);
    uint4* kdst = reinterpret_cast<uint4*>(&sK[m * HDIM]);
    uint4 vv[4];
    const uint4* vsrc = reinterpret_cast<const uint4*>(Vg + m * HDIM);
#pragma unroll
    for (int j = 0; j < 4; ++j) { kdst[j] = ksrc[j]; vv[j] = vsrc[j]; }
    const __bf16* vvb = reinterpret_cast<const __bf16*>(vv);
#pragma unroll
    for (int d = 0; d < HDIM; ++d) sVt[d * NPAD + m] = vvb[d];
  }
  __syncthreads();

  const int lane = tid & 31, wid = tid >> 5;
  const int half = lane >> 4, ln = lane & 15;
  __bf16* myP = &sP[wid][0];

  // All-ones B fragment: rowsum(P) = P x Ones via the matrix unit.
  const __bf16 one_bf = f2bf(1.0f);
  v16bf ones;
#pragma unroll
  for (int j = 0; j < 16; ++j) ones[j] = one_bf;

  for (int t = 0; t < 2; ++t) {
    const int mt = wid + 11 * t;  // query tile 0..21
    const __bf16* qrow = Qg + (mt * 16 + ln) * HDIM;
    v16bf qa = load_frag(qrow + 16 * half, qrow + 16 * half + 8);

    v8f o0 = v8f_zero(), o1 = v8f_zero();
    v8f lsum = v8f_zero();       // per-row softmax denominator (all 16 cols equal)
    float mrun[8];
#pragma unroll
    for (int i = 0; i < 8; ++i) mrun[i] = -3.0e38f;

    for (int jj = 0; jj < 11; ++jj) {
      const int m0 = jj * 32;
      const __bf16* k0 = &sK[(m0 + ln) * HDIM + half * 16];
      const __bf16* k1 = &sK[(m0 + 16 + ln) * HDIM + half * 16];
      v16bf kb0 = load_frag(k0, k0 + 8);
      v16bf kb1 = load_frag(k1, k1 + 8);
      v8f s0 = wmma_bf16(qa, kb0, v8f_zero());
      v8f s1 = wmma_bf16(qa, kb1, v8f_zero());

      const int col0 = m0 + ln, col1 = col0 + 16;
#pragma unroll
      for (int i = 0; i < 8; ++i) {
        const int n = mt * 16 + i + 8 * half;
        const float* brow = bh + n * NPAD;
        const float* mrow = mw + n * NPAD;
        float x0 = s0[i] + brow[col0] + mrow[col0];
        float x1 = s1[i] + brow[col1] + mrow[col1];
        float mx = red_max16(fmaxf(x0, x1));
        float mnew = fmaxf(mrun[i], mx);
        float alpha = __expf(mrun[i] - mnew);
        mrun[i] = mnew;
        float p0 = __expf(x0 - mnew), p1 = __expf(x1 - mnew);
        o0[i] *= alpha;
        o1[i] *= alpha;
        lsum[i] *= alpha;
        myP[(i + 8 * half) * 32 + ln] = f2bf(p0);
        myP[(i + 8 * half) * 32 + 16 + ln] = f2bf(p1);
      }
      // P back as A fragment (wave-local transpose through LDS)
      const __bf16* pr = &myP[ln * 32 + 16 * half];
      v16bf pa = load_frag(pr, pr + 8);
      const __bf16* v0 = &sVt[ln * NPAD + m0 + half * 16];
      const __bf16* v1 = &sVt[(16 + ln) * NPAD + m0 + half * 16];
      v16bf vb0 = load_frag(v0, v0 + 8);
      v16bf vb1 = load_frag(v1, v1 + 8);
      o0 = wmma_bf16(pa, vb0, o0);
      o1 = wmma_bf16(pa, vb1, o1);
      lsum = wmma_bf16(pa, ones, lsum);   // row-sum accumulate (denominator)
    }

#pragma unroll
    for (int i = 0; i < 8; ++i) {
      int n = mt * 16 + i + 8 * half;
      if (n < NTOK) {
        float inv = 1.0f / lsum[i];
        size_t base = ((size_t)b * NTOK + n) * CDIM + h * HDIM;
        O[base + ln] = f2bf(o0[i] * inv);
        O[base + 16 + ln] = f2bf(o1[i] * inv);
      }
    }
  }
}

// ---------------------------------------------------------------------------
// Output projection: out = O_bf16 @ proj_w^T + proj_b  (fp32 out).
// ---------------------------------------------------------------------------
__global__ __launch_bounds__(256) void gemm_proj_kernel(
    const __bf16* __restrict__ Oin, const __bf16* __restrict__ wbf,
    const float* __restrict__ bias, float* __restrict__ out) {
  __shared__ __bf16 sA[32 * CDIM];
  const int rb = blockIdx.x;
  const int tid = threadIdx.x;
  const uint4* src = reinterpret_cast<const uint4*>(Oin + (size_t)rb * 32 * CDIM);
  uint4* dst = reinterpret_cast<uint4*>(sA);
#pragma unroll
  for (int j = 0; j < 3; ++j) dst[tid + j * 256] = src[tid + j * 256];
  __syncthreads();

  const int lane = tid & 31, wid = tid >> 5;
  const int half = lane >> 4, ln = lane & 15;
  const int mt = wid & 1;
  const int ntbase = (wid >> 1) * 3;

  v8f acc[3];
#pragma unroll
  for (int nt = 0; nt < 3; ++nt) acc[nt] = v8f_zero();

  const __bf16* arow = &sA[(mt * 16 + ln) * CDIM];
#pragma unroll
  for (int k = 0; k < 6; ++k) {
    v16bf a = load_frag(arow + k * 32 + 16 * half,
                        arow + k * 32 + 16 * half + 8);
#pragma unroll
    for (int nt = 0; nt < 3; ++nt) {
      const __bf16* wrow =
          wbf + (size_t)((ntbase + nt) * 16 + ln) * CDIM + k * 32 + half * 16;
      v16bf b = load_frag(wrow, wrow + 8);
      acc[nt] = wmma_bf16(a, b, acc[nt]);
    }
  }

#pragma unroll
  for (int nt = 0; nt < 3; ++nt) {
    int col = (ntbase + nt) * 16 + ln;
    float bv = bias[col];
#pragma unroll
    for (int i = 0; i < 8; ++i) {
      int r = rb * 32 + mt * 16 + i + 8 * half;
      out[(size_t)r * CDIM + col] = acc[nt][i] + bv;
    }
  }
}

// ---------------------------------------------------------------------------
// Host launcher
// ---------------------------------------------------------------------------
extern "C" void kernel_launch(void* const* d_in, const int* in_sizes, int n_in,
                              void* d_out, int out_size, void* d_ws,
                              size_t ws_size, hipStream_t stream) {
  (void)in_sizes; (void)n_in; (void)out_size; (void)ws_size;
  const float* x_q    = (const float*)d_in[0];
  const float* x_kv   = (const float*)d_in[1];
  const float* mask   = (const float*)d_in[2];
  const float* q_w    = (const float*)d_in[3];
  const float* q_b    = (const float*)d_in[4];
  const float* kv_w   = (const float*)d_in[5];
  const float* kv_b   = (const float*)d_in[6];
  const float* proj_w = (const float*)d_in[7];
  const float* proj_b = (const float*)d_in[8];
  const float* rpb    = (const float*)d_in[9];
  const int*   rpi    = (const int*)d_in[10];

  char* ws = (char*)d_ws;
  size_t off = 0;
  auto take = [&](size_t bytes) {
    void* p = ws + off;
    off = (off + bytes + 255) & ~(size_t)255;
    return p;
  };
  __bf16* qwb   = (__bf16*)take((size_t)CDIM * CDIM * 2);        // 72 KB
  __bf16* kvwb  = (__bf16*)take((size_t)2 * CDIM * CDIM * 2);    // 144 KB
  __bf16* pwb   = (__bf16*)take((size_t)CDIM * CDIM * 2);        // 72 KB
  float*  biasD = (float*)take((size_t)HEADS * NPAD * NPAD * 4); // 2.97 MB
  float*  maskP = (float*)take((size_t)NWIN * NPAD * NPAD * 4);  // 31.7 MB
  size_t qkvBytes = (size_t)BATCH * HEADS * NPAD * HDIM * 2;     // 34.6 MB each
  __bf16* Qsb = (__bf16*)take(qkvBytes);
  __bf16* Ksb = (__bf16*)take(qkvBytes);
  __bf16* Vsb = (__bf16*)take(qkvBytes);
  __bf16* Ob  = (__bf16*)take((size_t)ROWS * CDIM * 2);          // 33.7 MB

  // Prep: weights -> bf16, dense bias + padded mask tables, zero pad rows.
  cvt_f32_to_bf16<<<(CDIM * CDIM + 255) / 256, 256, 0, stream>>>(q_w, qwb, CDIM * CDIM);
  cvt_f32_to_bf16<<<(2 * CDIM * CDIM + 255) / 256, 256, 0, stream>>>(kv_w, kvwb, 2 * CDIM * CDIM);
  cvt_f32_to_bf16<<<(CDIM * CDIM + 255) / 256, 256, 0, stream>>>(proj_w, pwb, CDIM * CDIM);
  build_bias<<<(HEADS * NPAD * NPAD + 255) / 256, 256, 0, stream>>>(rpb, rpi, biasD);
  build_maskP<<<(NWIN * NPAD * NPAD + 255) / 256, 256, 0, stream>>>(mask, maskP);
  zero_pads<<<(BATCH * HEADS * (NPAD - NTOK) * HDIM + 255) / 256, 256, 0, stream>>>(Qsb, Ksb, Vsb);

  // Projections (WMMA).
  gemm_qkv_kernel<3, 0><<<ROWBLOCKS, 256, 0, stream>>>(x_q, qwb, q_b, Qsb, nullptr);
  gemm_qkv_kernel<6, 1><<<ROWBLOCKS, 256, 0, stream>>>(x_kv, kvwb, kv_b, Ksb, Vsb);

  // Fused attention (flash-style, WMMA).
  attention_kernel<<<BATCH * HEADS, NPAD, 0, stream>>>(Qsb, Ksb, Vsb, biasD, maskP, Ob);

  // Output projection (WMMA) -> fp32.
  gemm_proj_kernel<<<ROWBLOCKS, 256, 0, stream>>>(Ob, pwb, proj_b, (float*)d_out);
}
